// IdpGANLayer_30760555774598
// MI455X (gfx1250) — matrix operand order repaired and checked by reading, not compiled, not run on hardware
//
#include <hip/hip_runtime.h>
#include <hip/hip_bf16.h>

typedef _Float16 half8 __attribute__((ext_vector_type(8)));
typedef _Float16 v16h  __attribute__((ext_vector_type(16)));
typedef float    v8f   __attribute__((ext_vector_type(8)));
typedef float    f4v   __attribute__((ext_vector_type(4)));
typedef int      i4v   __attribute__((ext_vector_type(4)));
typedef __attribute__((address_space(1))) i4v* gbl_i4p;  // global (AS1)
typedef __attribute__((address_space(3))) i4v* lds_i4p;  // LDS (AS3)

#define L_SEQ 1024
#define B_SZ  4
#define NIN   256
#define DM    256
#define NH    8
#define HDIM  32
#define IN2D  16

#if defined(__has_builtin)
#  if __has_builtin(__builtin_amdgcn_global_load_async_to_lds_b128)
#    define USE_ASYNC_LDS 1
#  endif
#endif

// ---------------- workspace layout (in _Float16 elements) ----------------
// sF   : fp16 copy of s,      (L*B, 256)                       1,048,576
// wT   : fp16 transposed Wq*scale, Wk, Wv, Wo  4 x (256x256)     262,144
// qF   : fp16 q, (B,H,L,HD)                                    1,048,576
// kF   : fp16 k, (B,H,L,HD)                                    1,048,576
// vT   : fp16 v, (B,H,HD,L)                                    1,048,576
// oF   : fp16 attention out, (L*B, 256)                        1,048,576
static constexpr size_t OFF_SF = 0;
static constexpr size_t OFF_WT = 1048576;
static constexpr size_t OFF_QF = 1310720;
static constexpr size_t OFF_KF = 2359296;
static constexpr size_t OFF_VT = 3407872;
static constexpr size_t OFF_OF = 4456448;

__device__ __forceinline__ v8f wmma16(v16h a, v16h b, v8f c) {
  // D = A(16x32 f16) x B(32x16 f16) + C(16x16 f32)
  return __builtin_amdgcn_wmma_f32_16x16x32_f16(false, a, false, b, (short)0, c,
                                                false, false);
}

// A-fragment (16x32, MxK) from row-major tile, row stride `ld` halves.
// lane<16: row=lane,   K = {0..7, 16..23}
// lane>=16: row=lane-16, K = {8..15, 24..31}
__device__ __forceinline__ v16h load_a16x32(const _Float16* base, int ld, int lane) {
  int row = lane & 15;
  int kb  = (lane < 16) ? 0 : 8;
  const _Float16* rp = base + row * ld + kb;
  half8 lo = *(const half8*)rp;
  half8 hi = *(const half8*)(rp + 16);
  v16h a;
#pragma unroll
  for (int t = 0; t < 8; ++t) { a[t] = lo[t]; a[8 + t] = hi[t]; }
  return a;
}

// B-fragment (32x16, KxN) where column n is contiguous over k with stride ldn:
// element (k, n) at base[n*ldn + k].
// lane<16: n=lane, K = {0..7, 16..23} ; lane>=16: n=lane-16, K = {8..15, 24..31}
__device__ __forceinline__ v16h load_b32x16(const _Float16* base, int ldn, int lane) {
  int n  = lane & 15;
  int k0 = (lane < 16) ? 0 : 8;
  const _Float16* cp = base + n * ldn + k0;
  half8 lo = *(const half8*)cp;
  half8 hi = *(const half8*)(cp + 16);
  v16h b;
#pragma unroll
  for (int t = 0; t < 8; ++t) { b[t] = lo[t]; b[8 + t] = hi[t]; }
  return b;
}

// ---------------- prep: fp32 -> fp16 conversions ----------------
__global__ void convert_s_kernel(const float* __restrict__ s,
                                 _Float16* __restrict__ sF) {
  int idx = blockIdx.x * blockDim.x + threadIdx.x;
  if (idx < L_SEQ * B_SZ * NIN) sF[idx] = (_Float16)s[idx];
}

// Build transposed fp16 weights: wt[m][n*256 + k] = W_m[k*256 + n]
// m=0 -> Wq scaled by 1/sqrt(D)=0.0625, m=1 Wk, m=2 Wv, m=3 Wo.
__global__ void convert_w_kernel(const float* __restrict__ Wq,
                                 const float* __restrict__ Wk,
                                 const float* __restrict__ Wv,
                                 const float* __restrict__ Wo,
                                 _Float16* __restrict__ wt) {
  int idx = blockIdx.x * blockDim.x + threadIdx.x;
  if (idx >= 4 * 65536) return;
  int m = idx >> 16;
  int e = idx & 65535;
  int n = e >> 8;
  int k = e & 255;
  const float* W = (m == 0) ? Wq : (m == 1) ? Wk : (m == 2) ? Wv : Wo;
  float val = W[k * 256 + n];
  if (m == 0) val *= 0.0625f;
  wt[idx] = (_Float16)val;
}

// ---------------- QKV projection: Y = X(4096x256) @ W(256x256) ----------------
// One wave per 16x16 output tile. Outputs written fp16 in attention layouts.
__global__ __launch_bounds__(32) void proj_kernel(
    const _Float16* __restrict__ sF, const _Float16* __restrict__ wT,
    _Float16* __restrict__ dq, _Float16* __restrict__ dk,
    _Float16* __restrict__ dv) {
  int r0   = blockIdx.x * 16;
  int c0   = blockIdx.y * 16;
  int mat  = blockIdx.z;   // 0=q 1=k 2=v
  int lane = threadIdx.x;

  const _Float16* w = wT + (size_t)mat * 65536;
  v8f acc = {};
#pragma unroll
  for (int kk = 0; kk < NIN; kk += 32) {
    v16h a = load_a16x32(sF + (size_t)r0 * NIN + kk, NIN, lane);
    v16h b = load_b32x16(w + (size_t)c0 * NIN + kk, NIN, lane);
    acc = wmma16(a, b, acc);
  }

  int col = c0 + (lane & 15);
  int h   = col >> 5;       // head
  int dl  = col & 31;       // dim within head
  int rowOff = (lane >> 4) * 8;
#pragma unroll
  for (int r = 0; r < 8; ++r) {
    int rowg = r0 + r + rowOff;        // row in (L*B), row = i*B + b
    int i = rowg >> 2;
    int b = rowg & 3;
    _Float16 val = (_Float16)acc[r];
    if (mat == 0)
      dq[(((size_t)(b * NH + h)) * L_SEQ + i) * HDIM + dl] = val;
    else if (mat == 1)
      dk[(((size_t)(b * NH + h)) * L_SEQ + i) * HDIM + dl] = val;
    else
      dv[(((size_t)(b * NH + h)) * HDIM + dl) * L_SEQ + i] = val;
  }
}

// ---------------- fused attention (flash-style, bias from p) ----------------
// Workgroup: (i-tile of 16 query rows, batch b); 8 waves = 8 heads.
// p tile staged once per key step in LDS, shared by all heads (p is head-free):
// p is the only O(L^2) HBM stream (256 MB) => this kernel is the roofline term.
__global__ __launch_bounds__(256) void attn_kernel(
    const _Float16* __restrict__ qF, const _Float16* __restrict__ kF,
    const _Float16* __restrict__ vT, const float* __restrict__ p,
    const float* __restrict__ W2d, const float* __restrict__ b2d,
    _Float16* __restrict__ oF) {
  __shared__ alignas(16) float     pTile[16 * 32 * IN2D];  // 32 KB
  __shared__ alignas(16) _Float16  pst[NH][16 * 32];       // 8 KB, per-wave P patch

  int i0   = blockIdx.x * 16;
  int b    = blockIdx.y;
  int tid  = threadIdx.x;
  int h    = tid >> 5;
  int lane = tid & 31;
  int bh   = b * NH + h;
  int colL   = lane & 15;
  int rowOff = (lane >> 4) * 8;

  float w2dv[IN2D];
#pragma unroll
  for (int c = 0; c < IN2D; ++c) w2dv[c] = W2d[h * IN2D + c];
  float bbias = b2d[h];

  const _Float16* qbase = qF + ((size_t)bh * L_SEQ + i0) * HDIM;
  v16h qfrag = load_a16x32(qbase, HDIM, lane);

  v8f acc0 = {}, acc1 = {};
  float mrow[8], lrow[8];
#pragma unroll
  for (int r = 0; r < 8; ++r) { mrow[r] = -1e30f; lrow[r] = 0.0f; }

  const float* pbase =
      p + (((size_t)b * L_SEQ + i0) * L_SEQ) * IN2D;  // p[b, i0, 0, 0]

  for (int j0 = 0; j0 < L_SEQ; j0 += 32) {
    __syncthreads();  // previous iteration fully consumed pTile / pst
    // stage p[b, i0..i0+15, j0..j0+31, :] -> LDS (read from HBM exactly once)
#ifdef USE_ASYNC_LDS
    for (int t = tid; t < 16 * 32 * (IN2D / 4); t += 256) {
      int e  = t >> 2;        // (i,j) pair 0..511
      int c4 = (t & 3) * 4;
      int il = e >> 5;
      int jl = e & 31;
      const float* gp =
          pbase + ((size_t)il * L_SEQ + (j0 + jl)) * IN2D + c4;
      __builtin_amdgcn_global_load_async_to_lds_b128(
          (gbl_i4p)gp, (lds_i4p)&pTile[e * IN2D + c4], 0, 0);
    }
#  if __has_builtin(__builtin_amdgcn_s_wait_asynccnt)
    __builtin_amdgcn_s_wait_asynccnt(0);
#  else
    asm volatile("s_wait_asynccnt 0x0" ::: "memory");
#  endif
#else
    for (int t = tid; t < 16 * 32 * (IN2D / 4); t += 256) {
      int e  = t >> 2;        // (i,j) pair 0..511
      int c4 = (t & 3) * 4;
      int il = e >> 5;
      int jl = e & 31;
      // p is streamed exactly once and exceeds L2 (256MB > 192MB): NT load
      f4v val = __builtin_nontemporal_load(
          (const f4v*)(pbase + ((size_t)il * L_SEQ + (j0 + jl)) * IN2D + c4));
      *(f4v*)(&pTile[e * IN2D + c4]) = val;
    }
#endif
    __syncthreads();

    // scores for 32 keys: two 16x16 tiles via WMMA (K = HD = 32)
    const _Float16* kb0 = kF + ((size_t)bh * L_SEQ + j0) * HDIM;
    v16h kf0 = load_b32x16(kb0, HDIM, lane);
    v16h kf1 = load_b32x16(kb0 + 16 * HDIM, HDIM, lane);
    v8f z = {};
    v8f s0 = wmma16(qfrag, kf0, z);
    v8f s1 = wmma16(qfrag, kf1, z);

    // bias + online softmax (C layout: row = r + rowOff, col = colL / colL+16)
#pragma unroll
    for (int r = 0; r < 8; ++r) {
      int row = r + rowOff;
      const f4v* pe0 = (const f4v*)&pTile[(row * 32 + colL) * IN2D];
      const f4v* pe1 = (const f4v*)&pTile[(row * 32 + colL + 16) * IN2D];
      float b0 = bbias, b1 = bbias;
#pragma unroll
      for (int cc = 0; cc < IN2D / 4; ++cc) {
        f4v x0 = pe0[cc];
        f4v x1 = pe1[cc];
#pragma unroll
        for (int u = 0; u < 4; ++u) {
          b0 += x0[u] * w2dv[cc * 4 + u];
          b1 += x1[u] * w2dv[cc * 4 + u];
        }
      }
      float sc0 = s0[r] + b0;
      float sc1 = s1[r] + b1;
      float rmax = fmaxf(sc0, sc1);
#pragma unroll
      for (int off = 1; off < 16; off <<= 1)
        rmax = fmaxf(rmax, __shfl_xor(rmax, off, 32));
      float mnew = fmaxf(mrow[r], rmax);
      float p0 = __expf(sc0 - mnew);
      float p1 = __expf(sc1 - mnew);
      float rsum = p0 + p1;
#pragma unroll
      for (int off = 1; off < 16; off <<= 1)
        rsum += __shfl_xor(rsum, off, 32);
      float alpha = __expf(mrow[r] - mnew);
      lrow[r] = lrow[r] * alpha + rsum;
      mrow[r] = mnew;
      acc0[r] *= alpha;
      acc1[r] *= alpha;
      pst[h][row * 32 + colL]      = (_Float16)p0;
      pst[h][row * 32 + colL + 16] = (_Float16)p1;
    }
    __syncthreads();  // pst visible for A-fragment reload

    // O += P(16x32) @ V(32x32), split into two 16-col WMMAs
    v16h pfrag = load_a16x32(&pst[h][0], 32, lane);
    const _Float16* vb = vT + (size_t)bh * HDIM * L_SEQ + j0;
    v16h vf0 = load_b32x16(vb, L_SEQ, lane);
    v16h vf1 = load_b32x16(vb + 16 * L_SEQ, L_SEQ, lane);
    acc0 = wmma16(pfrag, vf0, acc0);
    acc1 = wmma16(pfrag, vf1, acc1);
  }

  // normalize and emit fp16 o in (L, B, D) row-major (row = i*B + b)
#pragma unroll
  for (int r = 0; r < 8; ++r) {
    int row = r + rowOff;
    float inv = 1.0f / lrow[r];
    size_t o_off = (((size_t)(i0 + row)) * B_SZ + b) * DM + h * HDIM;
    oF[o_off + colL]      = (_Float16)(acc0[r] * inv);
    oF[o_off + colL + 16] = (_Float16)(acc1[r] * inv);
  }
}

// ---------------- output projection: out = oF @ Wo + bo (fp32 out) ----------------
__global__ __launch_bounds__(128) void out_kernel(
    const _Float16* __restrict__ oF, const _Float16* __restrict__ woT,
    const float* __restrict__ bo, float* __restrict__ out) {
  int r0   = blockIdx.x * 16;
  int wv   = threadIdx.x >> 5;
  int lane = threadIdx.x & 31;
  int c0   = (blockIdx.y * 4 + wv) * 16;

  v8f acc = {};
#pragma unroll
  for (int kk = 0; kk < DM; kk += 32) {
    v16h a = load_a16x32(oF + (size_t)r0 * DM + kk, DM, lane);
    v16h b = load_b32x16(woT + (size_t)c0 * DM + kk, DM, lane);
    acc = wmma16(a, b, acc);
  }

  int col = c0 + (lane & 15);
  int rowOff = (lane >> 4) * 8;
  float bias = bo[col];
#pragma unroll
  for (int r = 0; r < 8; ++r) {
    int rowg = r0 + r + rowOff;
    out[(size_t)rowg * NIN + col] = acc[r] + bias;
  }
}

extern "C" void kernel_launch(void* const* d_in, const int* in_sizes, int n_in,
                              void* d_out, int out_size, void* d_ws, size_t ws_size,
                              hipStream_t stream) {
  (void)in_sizes; (void)n_in; (void)out_size; (void)ws_size;
  const float* s   = (const float*)d_in[0];
  // d_in[1] (_k) and d_in[2] (_v) are unused by the reference (q,k,v all from s)
  const float* p   = (const float*)d_in[3];
  const float* Wq  = (const float*)d_in[4];
  const float* Wk  = (const float*)d_in[5];
  const float* Wv  = (const float*)d_in[6];
  const float* Wo  = (const float*)d_in[7];
  const float* bo  = (const float*)d_in[8];
  const float* W2d = (const float*)d_in[9];
  const float* b2d = (const float*)d_in[10];

  _Float16* wsH = (_Float16*)d_ws;

  convert_s_kernel<<<(L_SEQ * B_SZ * NIN) / 256, 256, 0, stream>>>(s, wsH + OFF_SF);
  convert_w_kernel<<<(4 * 65536) / 256, 256, 0, stream>>>(Wq, Wk, Wv, Wo,
                                                          wsH + OFF_WT);
  proj_kernel<<<dim3(256, 16, 3), 32, 0, stream>>>(
      wsH + OFF_SF, wsH + OFF_WT, wsH + OFF_QF, wsH + OFF_KF, wsH + OFF_VT);
  attn_kernel<<<dim3(L_SEQ / 16, B_SZ), 256, 0, stream>>>(
      wsH + OFF_QF, wsH + OFF_KF, wsH + OFF_VT, p, W2d, b2d, wsH + OFF_OF);
  out_kernel<<<dim3(256, 4), 128, 0, stream>>>(
      wsH + OFF_OF, wsH + OFF_WT + 3 * 65536, bo, (float*)d_out);
}
